// VLLMDualMLPAdapter_16441134809915
// MI455X (gfx1250) — matrix operand order repaired and checked by reading, not compiled, not run on hardware
//
#include <hip/hip_runtime.h>
#include <hip/hip_bf16.h>
#include <stdint.h>

// ---------------- problem constants ----------------
#define T_TOK 4096
#define H_DIM 2048
#define I_DIM 8192
#define A_NUM 8
#define N_DIM 512

// ---------------- tiling ----------------
#define BM 128           // row tile (tokens)
#define BN 64            // col tile
#define BK 32            // k tile (== WMMA K for bf16)
#define LDSK 48          // padded LDS row stride in bf16 elems (96B, 16B-aligned)

#define GLOBAL_AS __attribute__((address_space(1)))
#define LDS_AS    __attribute__((address_space(3)))

// 16-byte vector type matching the async-LDS builtin's parameter type
// ('__attribute__((__vector_size__(4 * sizeof(int)))) int' per clang diagnostic).
typedef int v4i_t __attribute__((vector_size(16)));

// CDNA5 async global->LDS path (ASYNCcnt), with safe fallback.
#if __has_builtin(__builtin_amdgcn_global_load_async_to_lds_b128) && \
    __has_builtin(__builtin_amdgcn_s_wait_asynccnt)
#define HAVE_ASYNC_LDS 1
#else
#define HAVE_ASYNC_LDS 0
#endif

#if __has_builtin(__builtin_amdgcn_cvt_pk_bf16_f32)
#define HAVE_PK_BF16 1
typedef __bf16 v2bf __attribute__((ext_vector_type(2)));
#else
#define HAVE_PK_BF16 0
#endif

typedef __bf16 v16bf __attribute__((ext_vector_type(16)));
typedef float  v8f   __attribute__((ext_vector_type(8)));

union FragU { v16bf v; uint4 q[2]; };

// fp32 -> bf16 round-to-nearest-even (scalar path / epilogues)
__device__ __forceinline__ unsigned short f2bf(float f) {
    unsigned int u = __builtin_bit_cast(unsigned int, f);
    unsigned int r = u + 0x7FFFu + ((u >> 16) & 1u);
    return (unsigned short)(r >> 16);
}

// 8 fp32 -> 8 bf16 packed in a uint4
__device__ __forceinline__ uint4 pack8_bf16(float4 f0, float4 f1) {
#if HAVE_PK_BF16
    v2bf a = __builtin_amdgcn_cvt_pk_bf16_f32(f0.x, f0.y);
    v2bf b = __builtin_amdgcn_cvt_pk_bf16_f32(f0.z, f0.w);
    v2bf c = __builtin_amdgcn_cvt_pk_bf16_f32(f1.x, f1.y);
    v2bf d = __builtin_amdgcn_cvt_pk_bf16_f32(f1.z, f1.w);
    uint4 q;
    q.x = __builtin_bit_cast(unsigned int, a);
    q.y = __builtin_bit_cast(unsigned int, b);
    q.z = __builtin_bit_cast(unsigned int, c);
    q.w = __builtin_bit_cast(unsigned int, d);
    return q;
#else
    union { unsigned short us[8]; uint4 q; } t;
    t.us[0] = f2bf(f0.x); t.us[1] = f2bf(f0.y); t.us[2] = f2bf(f0.z); t.us[3] = f2bf(f0.w);
    t.us[4] = f2bf(f1.x); t.us[5] = f2bf(f1.y); t.us[6] = f2bf(f1.z); t.us[7] = f2bf(f1.w);
    return t.q;
#endif
}

__device__ __forceinline__ void async_fence() {
#if HAVE_ASYNC_LDS
    __builtin_amdgcn_s_wait_asynccnt(0);
#endif
}

// ---- WMMA ----
__device__ __forceinline__ v8f wmma_bf16(v16bf a, v16bf b, v8f c) {
    return __builtin_amdgcn_wmma_f32_16x16x32_bf16(
        false, a, false, b, (short)0, c, false, false);
}

// ISA 16-bit A/B operand layout: lanes 0-15 hold row=lane (K 0-7, 16-23),
// lanes 16-31 hold row=lane-16 (K 8-15, 24-31). Two ds_load_b128 each.
__device__ __forceinline__ v16bf load_frag(const unsigned short* sBase, int rc, int lane) {
    int rr = rc + (lane & 15);
    int ko = (lane >> 4) * 8;
    FragU f;
    f.q[0] = *(const uint4*)&sBase[rr * LDSK + ko];
    f.q[1] = *(const uint4*)&sBase[rr * LDSK + ko + 16];
    return f.v;
}

// ---- staged B tile (fp32 global -> regs -> bf16 LDS) ----
struct BStage { float4 f0, f1; };

__device__ __forceinline__ BStage b_fetch(const float* __restrict__ src, int ldk, int k0, int tid) {
    int row = tid >> 2;          // 0..63
    int seg = tid & 3;           // 8 floats each
    const float4* p = (const float4*)(src + (size_t)row * ldk + k0 + seg * 8);
    BStage s; s.f0 = p[0]; s.f1 = p[1];
    return s;
}

__device__ __forceinline__ void b_commit(BStage s, unsigned short* sB, int tid) {
    int row = tid >> 2;
    int seg = tid & 3;
    *(uint4*)&sB[row * LDSK + seg * 8] = pack8_bf16(s.f0, s.f1);
}

// ---- staged/async A tile (bf16 global -> LDS), 32B per thread ----
struct AStage {
#if !HAVE_ASYNC_LDS
    uint4 q0, q1;
#endif
    int valid;
};

// issue: async path fires global_load_async_to_lds_b128 immediately (invalid rows
// zero-filled with ds stores); fallback stages into registers.
__device__ __forceinline__ AStage a_issue(const unsigned short* gp, unsigned short* lp, bool valid) {
    AStage st;
    st.valid = valid ? 1 : 0;
#if HAVE_ASYNC_LDS
    if (valid) {
        __builtin_amdgcn_global_load_async_to_lds_b128(
            (GLOBAL_AS v4i_t*)(uintptr_t)gp,
            (LDS_AS v4i_t*)(unsigned)(uintptr_t)lp, 0, 0);
        __builtin_amdgcn_global_load_async_to_lds_b128(
            (GLOBAL_AS v4i_t*)(uintptr_t)gp,
            (LDS_AS v4i_t*)(unsigned)(uintptr_t)lp, 16, 0);
    } else {
        uint4 z = {0u, 0u, 0u, 0u};
        *(uint4*)lp = z;
        *(uint4*)(lp + 8) = z;
    }
#else
    if (valid) {
        const uint4* p = (const uint4*)gp;
        st.q0 = p[0]; st.q1 = p[1];
    }
#endif
    return st;
}

__device__ __forceinline__ void a_commit(AStage st, unsigned short* lp) {
#if !HAVE_ASYNC_LDS
    if (st.valid) {
        *(uint4*)lp = st.q0;
        *(uint4*)(lp + 8) = st.q1;
    } else {
        uint4 z = {0u, 0u, 0u, 0u};
        *(uint4*)lp = z;
        *(uint4*)(lp + 8) = z;
    }
#endif
    (void)st; (void)lp;
}

// ---------------- kernel 1: x fp32 -> bf16 ----------------
__global__ __launch_bounds__(256) void convert_bf16_kernel(
        const float* __restrict__ src, unsigned short* __restrict__ dst, int n) {
    int i = (blockIdx.x * 256 + threadIdx.x) * 8;
    if (i + 8 <= n) {
        float4 f0 = *(const float4*)(src + i);
        float4 f1 = *(const float4*)(src + i + 4);
        *(uint4*)(dst + i) = pack8_bf16(f0, f1);
    }
}

// ---------------- kernels 2+3: bucket tokens by adapter ----------------
__global__ void zero_counts_kernel(int* counts) {
    if (threadIdx.x < A_NUM) counts[threadIdx.x] = 0;
}

__global__ __launch_bounds__(256) void bucket_kernel(
        const int* __restrict__ token_ids, int* __restrict__ counts, int* __restrict__ lists) {
    int t = blockIdx.x * 256 + threadIdx.x;
    if (t < T_TOK) {
        int a = token_ids[t];
        if (a >= 0) {
            int pos = atomicAdd(&counts[a], 1);
            lists[a * T_TOK + pos] = t;
        }
    }
}

// ---------------- kernel 4: base GEMM1 fused (inter = silu(x@Wg^T) * (x@Wu^T)) ----------------
__global__ __launch_bounds__(256) void gemm1_base_kernel(
        const unsigned short* __restrict__ xb,      // T x H bf16
        const float* __restrict__ wgu,              // 2I x H fp32
        unsigned short* __restrict__ inter) {       // T x I bf16
    __shared__ unsigned short sA[2][BM * LDSK];
    __shared__ unsigned short sBg[2][BN * LDSK];
    __shared__ unsigned short sBu[2][BN * LDSK];
    int tid = threadIdx.x;
    int nG = blockIdx.x * BN;
    int mG = blockIdx.y * BM;
    const float* wg = wgu + (size_t)nG * H_DIM;
    const float* wu = wgu + ((size_t)I_DIM + nG) * H_DIM;

    int row2 = tid >> 1, seg2 = tid & 1;
    const unsigned short* gpA = xb + ((size_t)(mG + row2)) * H_DIM + seg2 * 16;
    unsigned short* lpA[2] = { &sA[0][row2 * LDSK + seg2 * 16], &sA[1][row2 * LDSK + seg2 * 16] };

    int wave = tid >> 5, lane = tid & 31;
    int mb = (wave >> 1) * 32;
    int nb = (wave & 1) * 32;

    v8f accG[2][2], accU[2][2];
    v8f zero = {};
    for (int i = 0; i < 2; i++) for (int j = 0; j < 2; j++) { accG[i][j] = zero; accU[i][j] = zero; }

    // prologue: fill buffer 0
    AStage ast = a_issue(gpA, lpA[0], true);
    BStage bgs = b_fetch(wg, H_DIM, 0, tid);
    BStage bus = b_fetch(wu, H_DIM, 0, tid);
    a_commit(ast, lpA[0]);
    b_commit(bgs, sBg[0], tid);
    b_commit(bus, sBu[0], tid);
    async_fence();
    __syncthreads();

    const int KT = H_DIM / BK;
    for (int kt = 0; kt < KT; ++kt) {
        int cur = kt & 1, nxt = cur ^ 1;
        bool more = (kt + 1) < KT;
        int k1 = (kt + 1) * BK;
        if (more) {                                  // software pipeline: fetch k+1
            ast = a_issue(gpA + k1, lpA[nxt], true);
            bgs = b_fetch(wg, H_DIM, k1, tid);
            bus = b_fetch(wu, H_DIM, k1, tid);
        }
        v16bf aF[2], bG[2], bU[2];
        for (int mt = 0; mt < 2; ++mt) aF[mt] = load_frag(sA[cur], mb + mt * 16, lane);
        for (int nt = 0; nt < 2; ++nt) {
            bG[nt] = load_frag(sBg[cur], nb + nt * 16, lane);
            bU[nt] = load_frag(sBu[cur], nb + nt * 16, lane);
        }
        for (int mt = 0; mt < 2; ++mt)
            for (int nt = 0; nt < 2; ++nt) {
                accG[mt][nt] = wmma_bf16(aF[mt], bG[nt], accG[mt][nt]);
                accU[mt][nt] = wmma_bf16(aF[mt], bU[nt], accU[mt][nt]);
            }
        if (more) {
            a_commit(ast, lpA[nxt]);
            b_commit(bgs, sBg[nxt], tid);
            b_commit(bus, sBu[nxt], tid);
            async_fence();
        }
        __syncthreads();
    }

    for (int mt = 0; mt < 2; ++mt)
        for (int nt = 0; nt < 2; ++nt) {
            int n = nG + nb + nt * 16 + (lane & 15);
            int m0 = mG + mb + mt * 16 + ((lane >> 4) * 8);
            for (int v = 0; v < 8; ++v) {
                float g = accG[mt][nt][v];
                float u = accU[mt][nt][v];
                float h = (g / (1.f + __expf(-g))) * u;   // silu(g)*u
                inter[(size_t)(m0 + v) * I_DIM + n] = f2bf(h);
            }
        }
}

// ---------------- kernel 5: base GEMM2 (out = inter @ Wd^T) ----------------
__global__ __launch_bounds__(256) void gemm2_base_kernel(
        const unsigned short* __restrict__ inter,   // T x I bf16
        const float* __restrict__ wd,               // H x I fp32
        float* __restrict__ out) {                  // T x H fp32
    __shared__ unsigned short sA[2][BM * LDSK];
    __shared__ unsigned short sB[2][BN * LDSK];
    int tid = threadIdx.x;
    int nG = blockIdx.x * BN;
    int mG = blockIdx.y * BM;
    const float* wb = wd + (size_t)nG * I_DIM;

    int row2 = tid >> 1, seg2 = tid & 1;
    const unsigned short* gpA = inter + ((size_t)(mG + row2)) * I_DIM + seg2 * 16;
    unsigned short* lpA[2] = { &sA[0][row2 * LDSK + seg2 * 16], &sA[1][row2 * LDSK + seg2 * 16] };

    int wave = tid >> 5, lane = tid & 31;
    int mb = (wave >> 1) * 32;
    int nb = (wave & 1) * 32;

    v8f acc[2][2];
    v8f zero = {};
    for (int i = 0; i < 2; i++) for (int j = 0; j < 2; j++) acc[i][j] = zero;

    AStage ast = a_issue(gpA, lpA[0], true);
    BStage bs = b_fetch(wb, I_DIM, 0, tid);
    a_commit(ast, lpA[0]);
    b_commit(bs, sB[0], tid);
    async_fence();
    __syncthreads();

    const int KT = I_DIM / BK;
    for (int kt = 0; kt < KT; ++kt) {
        int cur = kt & 1, nxt = cur ^ 1;
        bool more = (kt + 1) < KT;
        int k1 = (kt + 1) * BK;
        if (more) {
            ast = a_issue(gpA + k1, lpA[nxt], true);
            bs = b_fetch(wb, I_DIM, k1, tid);
        }
        v16bf aF[2], bF[2];
        for (int mt = 0; mt < 2; ++mt) aF[mt] = load_frag(sA[cur], mb + mt * 16, lane);
        for (int nt = 0; nt < 2; ++nt) bF[nt] = load_frag(sB[cur], nb + nt * 16, lane);
        for (int mt = 0; mt < 2; ++mt)
            for (int nt = 0; nt < 2; ++nt)
                acc[mt][nt] = wmma_bf16(aF[mt], bF[nt], acc[mt][nt]);
        if (more) {
            a_commit(ast, lpA[nxt]);
            b_commit(bs, sB[nxt], tid);
            async_fence();
        }
        __syncthreads();
    }

    for (int mt = 0; mt < 2; ++mt)
        for (int nt = 0; nt < 2; ++nt) {
            int n = nG + nb + nt * 16 + (lane & 15);
            int m0 = mG + mb + mt * 16 + ((lane >> 4) * 8);
            for (int v = 0; v < 8; ++v)
                out[(size_t)(m0 + v) * H_DIM + n] = acc[mt][nt][v];
        }
}

// ---------------- kernel 6: adapter gate/up (token-gathered, fused silu) ----------------
__global__ __launch_bounds__(256) void adapter_gu_kernel(
        const unsigned short* __restrict__ xb,      // T x H bf16
        const float* __restrict__ gate_w,           // A x N x H fp32
        const float* __restrict__ up_w,             // A x N x H fp32
        const int* __restrict__ lists,
        const int* __restrict__ counts,
        unsigned short* __restrict__ out_inter) {   // T x N bf16
    __shared__ unsigned short sA[2][BM * LDSK];
    __shared__ unsigned short sBg[2][BN * LDSK];
    __shared__ unsigned short sBu[2][BN * LDSK];
    __shared__ int toks[BM];

    int a = blockIdx.z;
    int cnt = counts[a];
    int rbase = blockIdx.y * BM;
    if (rbase >= cnt) return;                       // grouped-GEMM early exit
    int tid = threadIdx.x;
    if (tid < BM) {
        int idx = rbase + tid;
        toks[tid] = (idx < cnt) ? lists[a * T_TOK + idx] : -1;
    }
    __syncthreads();

    int nG = blockIdx.x * BN;
    const float* wg = gate_w + (size_t)a * N_DIM * H_DIM + (size_t)nG * H_DIM;
    const float* wu = up_w   + (size_t)a * N_DIM * H_DIM + (size_t)nG * H_DIM;

    int row2 = tid >> 1, seg2 = tid & 1;
    int tokRow = toks[row2];
    bool vA = tokRow >= 0;
    const unsigned short* gpA = xb + (size_t)(vA ? tokRow : 0) * H_DIM + seg2 * 16;
    unsigned short* lpA[2] = { &sA[0][row2 * LDSK + seg2 * 16], &sA[1][row2 * LDSK + seg2 * 16] };

    int wave = tid >> 5, lane = tid & 31;
    int mb = (wave >> 1) * 32;
    int nb = (wave & 1) * 32;

    v8f accG[2][2], accU[2][2];
    v8f zero = {};
    for (int i = 0; i < 2; i++) for (int j = 0; j < 2; j++) { accG[i][j] = zero; accU[i][j] = zero; }

    AStage ast = a_issue(gpA, lpA[0], vA);
    BStage bgs = b_fetch(wg, H_DIM, 0, tid);
    BStage bus = b_fetch(wu, H_DIM, 0, tid);
    a_commit(ast, lpA[0]);
    b_commit(bgs, sBg[0], tid);
    b_commit(bus, sBu[0], tid);
    async_fence();
    __syncthreads();

    const int KT = H_DIM / BK;
    for (int kt = 0; kt < KT; ++kt) {
        int cur = kt & 1, nxt = cur ^ 1;
        bool more = (kt + 1) < KT;
        int k1 = (kt + 1) * BK;
        if (more) {
            ast = a_issue(gpA + k1, lpA[nxt], vA);
            bgs = b_fetch(wg, H_DIM, k1, tid);
            bus = b_fetch(wu, H_DIM, k1, tid);
        }
        v16bf aF[2], bG[2], bU[2];
        for (int mt = 0; mt < 2; ++mt) aF[mt] = load_frag(sA[cur], mb + mt * 16, lane);
        for (int nt = 0; nt < 2; ++nt) {
            bG[nt] = load_frag(sBg[cur], nb + nt * 16, lane);
            bU[nt] = load_frag(sBu[cur], nb + nt * 16, lane);
        }
        for (int mt = 0; mt < 2; ++mt)
            for (int nt = 0; nt < 2; ++nt) {
                accG[mt][nt] = wmma_bf16(aF[mt], bG[nt], accG[mt][nt]);
                accU[mt][nt] = wmma_bf16(aF[mt], bU[nt], accU[mt][nt]);
            }
        if (more) {
            a_commit(ast, lpA[nxt]);
            b_commit(bgs, sBg[nxt], tid);
            b_commit(bus, sBu[nxt], tid);
            async_fence();
        }
        __syncthreads();
    }

    for (int mt = 0; mt < 2; ++mt)
        for (int nt = 0; nt < 2; ++nt) {
            int n = nG + nb + nt * 16 + (lane & 15);
            int r0 = mb + mt * 16 + ((lane >> 4) * 8);
            for (int v = 0; v < 8; ++v) {
                int tok = toks[r0 + v];
                if (tok >= 0) {
                    float g = accG[mt][nt][v];
                    float u = accU[mt][nt][v];
                    float h = (g / (1.f + __expf(-g))) * u;
                    out_inter[(size_t)tok * N_DIM + n] = f2bf(h);
                }
            }
        }
}

// ---------------- kernel 7: adapter down + scaled accumulate into out ----------------
__global__ __launch_bounds__(256) void adapter_down_kernel(
        const unsigned short* __restrict__ rint,    // T x N bf16
        const unsigned short* __restrict__ fint,    // T x N bf16
        const float* __restrict__ rd,               // A x H x N fp32
        const float* __restrict__ fd,               // A x H x N fp32
        const int* __restrict__ lists,
        const int* __restrict__ counts,
        const float* __restrict__ scales,           // A x 2 fp32
        float* __restrict__ out) {                  // T x H fp32 (already holds base_out)
    __shared__ unsigned short sAr[BM * LDSK];
    __shared__ unsigned short sAf[BM * LDSK];
    __shared__ unsigned short sBr[BN * LDSK];
    __shared__ unsigned short sBf[BN * LDSK];
    __shared__ int toks[BM];

    int a = blockIdx.z;
    int cnt = counts[a];
    int rbase = blockIdx.y * BM;
    if (rbase >= cnt) return;
    int tid = threadIdx.x;
    if (tid < BM) {
        int idx = rbase + tid;
        toks[tid] = (idx < cnt) ? lists[a * T_TOK + idx] : -1;
    }
    __syncthreads();

    int nG = blockIdx.x * BN;
    const float* wr = rd + (size_t)a * H_DIM * N_DIM + (size_t)nG * N_DIM;
    const float* wf = fd + (size_t)a * H_DIM * N_DIM + (size_t)nG * N_DIM;
    float s0 = scales[a * 2 + 0];
    float s1 = scales[a * 2 + 1];

    int row2 = tid >> 1, seg2 = tid & 1;
    int tokRow = toks[row2];
    bool vA = tokRow >= 0;
    const unsigned short* gpR = rint + (size_t)(vA ? tokRow : 0) * N_DIM + seg2 * 16;
    const unsigned short* gpF = fint + (size_t)(vA ? tokRow : 0) * N_DIM + seg2 * 16;
    unsigned short* lpR = &sAr[row2 * LDSK + seg2 * 16];
    unsigned short* lpF = &sAf[row2 * LDSK + seg2 * 16];

    int wave = tid >> 5, lane = tid & 31;
    int mb = (wave >> 1) * 32;
    int nb = (wave & 1) * 32;

    v8f accR[2][2], accF[2][2];
    v8f zero = {};
    for (int i = 0; i < 2; i++) for (int j = 0; j < 2; j++) { accR[i][j] = zero; accF[i][j] = zero; }

    for (int k0 = 0; k0 < N_DIM; k0 += BK) {
        __syncthreads();
        AStage ar = a_issue(gpR + k0, lpR, vA);
        AStage af = a_issue(gpF + k0, lpF, vA);
        BStage br = b_fetch(wr, N_DIM, k0, tid);
        BStage bf = b_fetch(wf, N_DIM, k0, tid);
        a_commit(ar, lpR);
        a_commit(af, lpF);
        b_commit(br, sBr, tid);
        b_commit(bf, sBf, tid);
        async_fence();
        __syncthreads();
        v16bf aR[2], aX[2], bR[2], bX[2];
        for (int mt = 0; mt < 2; ++mt) {
            aR[mt] = load_frag(sAr, mb + mt * 16, lane);
            aX[mt] = load_frag(sAf, mb + mt * 16, lane);
        }
        for (int nt = 0; nt < 2; ++nt) {
            bR[nt] = load_frag(sBr, nb + nt * 16, lane);
            bX[nt] = load_frag(sBf, nb + nt * 16, lane);
        }
        for (int mt = 0; mt < 2; ++mt)
            for (int nt = 0; nt < 2; ++nt) {
                accR[mt][nt] = wmma_bf16(aR[mt], bR[nt], accR[mt][nt]);
                accF[mt][nt] = wmma_bf16(aX[mt], bX[nt], accF[mt][nt]);
            }
    }
    for (int mt = 0; mt < 2; ++mt)
        for (int nt = 0; nt < 2; ++nt) {
            int n = nG + nb + nt * 16 + (lane & 15);
            int r0 = mb + mt * 16 + ((lane >> 4) * 8);
            for (int v = 0; v < 8; ++v) {
                int tok = toks[r0 + v];
                if (tok >= 0) {
                    size_t o = (size_t)tok * H_DIM + n;   // unique writer per (tok, n)
                    out[o] = out[o] + s0 * accR[mt][nt][v] + s1 * accF[mt][nt][v];
                }
            }
        }
}

// ---------------- workspace layout (bytes), needs ~92.5 MB ----------------
#define OFF_XB     ((size_t)0)
#define OFF_INTER  ((size_t)T_TOK * H_DIM * 2)
#define OFF_RINT   (OFF_INTER + (size_t)T_TOK * I_DIM * 2)
#define OFF_FINT   (OFF_RINT + (size_t)T_TOK * N_DIM * 2)
#define OFF_CNT    (OFF_FINT + (size_t)T_TOK * N_DIM * 2)
#define OFF_LISTS  (OFF_CNT + 1024)

extern "C" void kernel_launch(void* const* d_in, const int* in_sizes, int n_in,
                              void* d_out, int out_size, void* d_ws, size_t ws_size,
                              hipStream_t stream) {
    const float* x           = (const float*)d_in[0];
    const int*   token_ids   = (const int*)d_in[1];
    const float* w_gate_up   = (const float*)d_in[2];
    const float* w_down      = (const float*)d_in[3];
    const float* retain_gate = (const float*)d_in[4];
    const float* retain_up   = (const float*)d_in[5];
    const float* retain_down = (const float*)d_in[6];
    const float* forget_gate = (const float*)d_in[7];
    const float* forget_up   = (const float*)d_in[8];
    const float* forget_down = (const float*)d_in[9];
    const float* scales      = (const float*)d_in[10];
    float* out = (float*)d_out;

    char* ws = (char*)d_ws;
    unsigned short* xb    = (unsigned short*)(ws + OFF_XB);
    unsigned short* inter = (unsigned short*)(ws + OFF_INTER);
    unsigned short* rint  = (unsigned short*)(ws + OFF_RINT);
    unsigned short* fint  = (unsigned short*)(ws + OFF_FINT);
    int* counts = (int*)(ws + OFF_CNT);
    int* lists  = (int*)(ws + OFF_LISTS);

    {   // 1) x -> bf16
        int n = T_TOK * H_DIM;
        convert_bf16_kernel<<<n / (8 * 256), 256, 0, stream>>>(x, xb, n);
    }
    // 2+3) bucket tokens by adapter
    zero_counts_kernel<<<1, 32, 0, stream>>>(counts);
    bucket_kernel<<<T_TOK / 256, 256, 0, stream>>>(token_ids, counts, lists);

    // 4) base GEMM1 fused silu-gate -> inter (bf16)
    gemm1_base_kernel<<<dim3(I_DIM / BN, T_TOK / BM), 256, 0, stream>>>(xb, w_gate_up, inter);

    // 5) base GEMM2 -> out (fp32, full coverage)
    gemm2_base_kernel<<<dim3(H_DIM / BN, T_TOK / BM), 256, 0, stream>>>(inter, w_down, out);

    // 6) adapter gate/up for both branches (token-gathered grouped GEMM)
    adapter_gu_kernel<<<dim3(N_DIM / BN, T_TOK / BM, A_NUM), 256, 0, stream>>>(
        xb, retain_gate, retain_up, lists, counts, rint);
    adapter_gu_kernel<<<dim3(N_DIM / BN, T_TOK / BM, A_NUM), 256, 0, stream>>>(
        xb, forget_gate, forget_up, lists, counts, fint);

    // 7) adapter down + scaled accumulate into out
    adapter_down_kernel<<<dim3(H_DIM / BN, T_TOK / BM, A_NUM), 256, 0, stream>>>(
        rint, fint, retain_down, forget_down, lists, counts, scales, out);
}
